// GIN_44100724196039
// MI455X (gfx1250) — matrix operand order repaired and checked by reading, not compiled, not run on hardware
//
#include <hip/hip_runtime.h>
#include <hip/hip_bf16.h>

#define N_NODES 50000
#define N_EDGES 800000
#define NUM_FEATS 128
#define DIM 256
#define NUM_GRAPHS 256
#define NUM_CLASSES 16
#define BN_EPS 1e-5f

typedef __attribute__((ext_vector_type(16))) __bf16 v16bf;
typedef __attribute__((ext_vector_type(8)))  float  v8f;
typedef __attribute__((ext_vector_type(4)))  unsigned int v4u;
typedef __attribute__((ext_vector_type(8)))  int  v8i;
typedef __attribute__((ext_vector_type(4)))  int  v4i;

__device__ __forceinline__ unsigned short f2bf(float f) {
    unsigned int u = __float_as_uint(f);
    unsigned int r = u + 0x7FFFu + ((u >> 16) & 1u);   // round-to-nearest-even
    return (unsigned short)(r >> 16);
}

// ---------------- weight convert + transpose: W[K][N] f32 -> WT[N][K] bf16 --------
__global__ void convT_kernel(const float* __restrict__ W, unsigned short* __restrict__ WT,
                             int K, int N) {
    int idx = blockIdx.x * blockDim.x + threadIdx.x;
    if (idx >= K * N) return;
    int k = idx / N, n = idx % N;
    WT[(size_t)n * K + k] = f2bf(W[idx]);
}

// ---------------- flat f32 -> bf16 (float4 granularity) --------------------------
__global__ void f2bf_kernel(const float* __restrict__ in, unsigned short* __restrict__ out, int n4) {
    int i = blockIdx.x * blockDim.x + threadIdx.x;
    if (i >= n4) return;
    float4 v = ((const float4*)in)[i];
    ushort4 o;
    o.x = f2bf(v.x); o.y = f2bf(v.y); o.z = f2bf(v.z); o.w = f2bf(v.w);
    ((ushort4*)out)[i] = o;
}

// ---------------- copy f32 (float4 granularity) ----------------------------------
__global__ void copy4_kernel(const float* __restrict__ src, float* __restrict__ dst, int n4) {
    int i = blockIdx.x * blockDim.x + threadIdx.x;
    if (i >= n4) return;
    ((float4*)dst)[i] = ((const float4*)src)[i];
}

// ---------------- zero f32 -------------------------------------------------------
__global__ void zero_kernel(float* __restrict__ p, int n) {
    int i = blockIdx.x * blockDim.x + threadIdx.x;
    if (i < n) p[i] = 0.0f;
}

// ---------------- edge scatter-add: z[dst] += h[src]  (L2-resident atomics) ------
__global__ void edge_agg_kernel(const float* __restrict__ h, const int* __restrict__ src,
                                const int* __restrict__ dst, float* __restrict__ z,
                                int E, int D) {
    int per = D >> 2;                          // float4 chunks per edge
    int tid = blockIdx.x * blockDim.x + threadIdx.x;
    if (tid >= E * per) return;
    int e  = tid / per;
    int f4 = (tid % per) << 2;
    int s = src[e], d = dst[e];
    float4 v = *(const float4*)(h + (size_t)s * D + f4);
    float* p = z + (size_t)d * D + f4;
    atomicAdd(p + 0, v.x);
    atomicAdd(p + 1, v.y);
    atomicAdd(p + 2, v.z);
    atomicAdd(p + 3, v.w);
}

// =================================================================================
// WMMA bf16 GEMM: C[M][N] = relu?(A[M][K] * W[K][N] + bias)
//  - Abf: activations, bf16 [M][K] row-major (pre-converted)
//  - WT : weights,     bf16 [N][K] row-major (pre-transposed)
//  - A tile (32 x K) staged global->LDS by the Tensor Data Mover (one 2D D# per
//    block; TDM zero-fills rows past M via tensor_dim1 = M - row0)
//  - B fragments loaded straight from global (weights are L2-resident)
//  - Block: 128 threads = 4 waves; block tile 32 x 128; per-wave 32 x 32
//    (2x2 register tile -> each fragment feeds 2 WMMAs)
// =================================================================================
__global__ __launch_bounds__(128)
void gemm_wmma_kernel(const unsigned short* __restrict__ Abf,
                      const unsigned short* __restrict__ WT,
                      const float* __restrict__ bias, void* __restrict__ Cout,
                      int M, int N, int K, int relu, int store_bf16) {
    __shared__ alignas(16) unsigned short lA[32 * 256];   // 32 rows x K (K<=256) bf16

    const int tid  = threadIdx.x;
    const int row0 = blockIdx.y * 32;
    const int col0 = blockIdx.x * 128;

    // ---- TDM stage of the A tile: wave 0 issues one tensor_load_to_lds ----
    if (tid < 32) {
        unsigned      lds_off = (unsigned)(size_t)&lA[0];          // LDS aperture: low 32b = LDS offset
        unsigned long long ga = (unsigned long long)(size_t)(Abf + (size_t)row0 * K);
        unsigned td0 = (unsigned)K;            // tensor_dim0 (elements)
        unsigned td1 = (unsigned)(M - row0);   // remaining rows -> TDM zero-fills tile OOB
        unsigned long long st0 = (unsigned long long)K;  // tensor_dim0_stride (elements)

        v4u g0;
        g0.x = 1u;                                               // count=1, user descriptor
        g0.y = lds_off;                                          // lds_addr
        g0.z = (unsigned)(ga & 0xFFFFFFFFull);                   // global_addr[31:0]
        g0.w = (unsigned)((ga >> 32) & 0x01FFFFFFull) | 0x80000000u;  // addr[56:32] | type=2

        v8i g1;
        g1[0] = (int)(1u << 16);                                 // data_size=1 (2 bytes)
        g1[1] = (int)((td0 & 0xFFFFu) << 16);                    // tensor_dim0[15:0]
        g1[2] = (int)((td0 >> 16) | ((td1 & 0xFFFFu) << 16));    // td0[31:16] | td1[15:0]
        g1[3] = (int)((td1 >> 16) | ((unsigned)K << 16));        // td1[31:16] | tile_dim0=K
        g1[4] = (int)32u;                                        // tile_dim1=32, tile_dim2=0
        g1[5] = (int)(unsigned)(st0 & 0xFFFFFFFFull);            // stride0[31:0]
        g1[6] = (int)(unsigned)(st0 >> 32);                      // stride0[47:32] | stride1[15:0]=0
        g1[7] = 0;                                               // stride1[47:16]=0

        v4i g2 = {0, 0, 0, 0};
        v4i g3 = {0, 0, 0, 0};
#if __clang_major__ >= 23
        v8i g4 = {0, 0, 0, 0, 0, 0, 0, 0};
        __builtin_amdgcn_tensor_load_to_lds(g0, g1, g2, g3, g4, 0);
#else
        __builtin_amdgcn_tensor_load_to_lds(g0, g1, g2, g3, 0);
#endif
        __builtin_amdgcn_s_wait_tensorcnt(0);
    }
    __syncthreads();

    const int lane  = tid & 31;
    const int wave  = tid >> 5;
    const int lid16 = lane & 15;
    const int hi    = lane >> 4;           // 0 = lanes 0-15, 1 = lanes 16-31
    const int khA   = hi * 8;              // A striping: {0..7,16..23} / {8..15,24..31}
    const int khB   = hi * 8;              // B striping: {0..7,16..23} / {8..15,24..31}

    union Frag { v16bf v; uint4 q[2]; };

    const unsigned short* lAr0 = &lA[(size_t)lid16 * K];          // m-tile 0 row
    const unsigned short* lAr1 = &lA[(size_t)(16 + lid16) * K];   // m-tile 1 row

    const int n0  = col0 + wave * 32 + lid16;
    const int n1  = n0 + 16;
    const bool ok0 = n0 < N, ok1 = n1 < N;
    const unsigned short* bp0 = WT + (size_t)(ok0 ? n0 : 0) * K;
    const unsigned short* bp1 = WT + (size_t)(ok1 ? n1 : 0) * K;

    v8f acc00 = {}, acc01 = {}, acc10 = {}, acc11 = {};

    for (int ks = 0; ks < K; ks += 32) {
        Frag fa0, fa1, fb0, fb1;
        fa0.q[0] = *(const uint4*)(lAr0 + ks + khA);
        fa0.q[1] = *(const uint4*)(lAr0 + ks + khA + 16);
        fa1.q[0] = *(const uint4*)(lAr1 + ks + khA);
        fa1.q[1] = *(const uint4*)(lAr1 + ks + khA + 16);
        fb0.q[0] = *(const uint4*)(bp0 + ks + khB);
        fb0.q[1] = *(const uint4*)(bp0 + ks + khB + 16);
        fb1.q[0] = *(const uint4*)(bp1 + ks + khB);
        fb1.q[1] = *(const uint4*)(bp1 + ks + khB + 16);
        acc00 = __builtin_amdgcn_wmma_f32_16x16x32_bf16(false, fa0.v, false, fb0.v, (short)0, acc00, false, false);
        acc01 = __builtin_amdgcn_wmma_f32_16x16x32_bf16(false, fa0.v, false, fb1.v, (short)0, acc01, false, false);
        acc10 = __builtin_amdgcn_wmma_f32_16x16x32_bf16(false, fa1.v, false, fb0.v, (short)0, acc10, false, false);
        acc11 = __builtin_amdgcn_wmma_f32_16x16x32_bf16(false, fa1.v, false, fb1.v, (short)0, acc11, false, false);
    }

    // ---- epilogue: C layout per tile: VGPR v -> row v + 8*hi, col = lane&15 ----
    float b0 = ok0 ? bias[n0] : 0.0f;
    float b1 = ok1 ? bias[n1] : 0.0f;
    float*          Cf = (float*)Cout;
    unsigned short* Cb = (unsigned short*)Cout;

#pragma unroll
    for (int mt = 0; mt < 2; ++mt) {
        const v8f a0 = mt ? acc10 : acc00;
        const v8f a1 = mt ? acc11 : acc01;
#pragma unroll
        for (int v = 0; v < 8; ++v) {
            int r = row0 + mt * 16 + v + hi * 8;
            if (r < M) {
                if (ok0) {
                    float val = a0[v] + b0;
                    if (relu) val = fmaxf(val, 0.0f);
                    if (store_bf16) Cb[(size_t)r * N + n0] = f2bf(val);
                    else            Cf[(size_t)r * N + n0] = val;
                }
                if (ok1) {
                    float val = a1[v] + b1;
                    if (relu) val = fmaxf(val, 0.0f);
                    if (store_bf16) Cb[(size_t)r * N + n1] = f2bf(val);
                    else            Cf[(size_t)r * N + n1] = val;
                }
            }
        }
    }
}

// ---------------- BN stats: per-channel sum / sumsq via atomics ------------------
__global__ void bn_stats_kernel(const float* __restrict__ z, float* __restrict__ stats, int M) {
    const int c    = threadIdx.x;              // channel 0..255
    const int row0 = blockIdx.x * 128;
    float s = 0.f, s2 = 0.f;
    int rend = min(row0 + 128, M);
    for (int r = row0; r < rend; ++r) {
        float v = z[(size_t)r * DIM + c];
        s += v; s2 += v * v;
    }
    atomicAdd(&stats[c], s);
    atomicAdd(&stats[DIM + c], s2);
}

// ---------------- BN apply -------------------------------------------------------
__global__ void bn_apply_kernel(const float* __restrict__ z, const float* __restrict__ stats,
                                const float* __restrict__ gamma, const float* __restrict__ beta,
                                float* __restrict__ out, int M) {
    const int c = threadIdx.x;
    const int r = blockIdx.x;
    float inv_m = 1.0f / (float)M;
    float mean  = stats[c] * inv_m;
    float var   = stats[DIM + c] * inv_m - mean * mean;
    float scale = gamma[c] * rsqrtf(var + BN_EPS);
    size_t idx = (size_t)r * DIM + c;
    out[idx] = scale * (z[idx] - mean) + beta[c];
}

// ---------------- graph pooling: pooled[batch[n]] += h[n] ------------------------
__global__ void pool_kernel(const float* __restrict__ h, const int* __restrict__ batch,
                            float* __restrict__ pooled) {
    const int c = threadIdx.x;
    const int n = blockIdx.x;
    int g = batch[n];
    atomicAdd(&pooled[(size_t)g * DIM + c], h[(size_t)n * DIM + c]);
}

static inline int cdiv(int a, int b) { return (a + b - 1) / b; }

extern "C" void kernel_launch(void* const* d_in, const int* in_sizes, int n_in,
                              void* d_out, int out_size, void* d_ws, size_t ws_size,
                              hipStream_t stream) {
    // ---- inputs (setup_inputs order) ----
    const float* x    = (const float*)d_in[0];
    const int*   ei   = (const int*)d_in[1];
    const int*   batch= (const int*)d_in[2];
    const float* w1a = (const float*)d_in[3];  const float* b1a = (const float*)d_in[4];
    const float* w1b = (const float*)d_in[5];  const float* b1b = (const float*)d_in[6];
    const float* g1  = (const float*)d_in[7];  const float* be1 = (const float*)d_in[8];
    const float* w2a = (const float*)d_in[9];  const float* b2a = (const float*)d_in[10];
    const float* w2b = (const float*)d_in[11]; const float* b2b = (const float*)d_in[12];
    const float* g2  = (const float*)d_in[13]; const float* be2 = (const float*)d_in[14];
    const float* w3a = (const float*)d_in[15]; const float* b3a = (const float*)d_in[16];
    const float* w3b = (const float*)d_in[17]; const float* b3b = (const float*)d_in[18];
    const float* g3  = (const float*)d_in[19]; const float* be3 = (const float*)d_in[20];
    const float* fw1 = (const float*)d_in[21]; const float* fb1 = (const float*)d_in[22];
    const float* fw2 = (const float*)d_in[23]; const float* fb2 = (const float*)d_in[24];

    const int* src = ei;              // edge_index[0]
    const int* dst = ei + N_EDGES;    // edge_index[1]

    // ---- workspace layout ----
    const size_t ND = (size_t)N_NODES * DIM;
    float* zsum   = (float*)d_ws;               // [N_NODES, DIM] f32 (layer-1 uses it packed as [N_NODES, 128])
    float* t1     = zsum + ND;                  // aliased as bf16 [N_NODES, DIM] GEMM1 output
    float* hbuf   = t1 + ND;                    // [N_NODES, DIM] f32 (BN output / layer input)
    float* stats  = hbuf + ND;                  // [512]
    float* pooled = stats + 512;                // [256, 256] f32
    float* headh  = pooled + (size_t)NUM_GRAPHS * DIM;  // aliased as bf16 [256, 256]
    unsigned short* w1aT = (unsigned short*)(headh + (size_t)NUM_GRAPHS * DIM);
    unsigned short* w1bT = w1aT + (size_t)DIM * NUM_FEATS;
    unsigned short* w2aT = w1bT + (size_t)DIM * DIM;
    unsigned short* w2bT = w2aT + (size_t)DIM * DIM;
    unsigned short* w3aT = w2bT + (size_t)DIM * DIM;
    unsigned short* w3bT = w3aT + (size_t)DIM * DIM;
    unsigned short* fw1T = w3bT + (size_t)DIM * DIM;
    unsigned short* fw2T = fw1T + (size_t)DIM * DIM;
    unsigned short* abf  = fw2T + (size_t)DIM * NUM_CLASSES;  // bf16 activation staging [N_NODES, DIM]

    unsigned short* t1bf    = (unsigned short*)t1;
    unsigned short* headhbf = (unsigned short*)headh;

    // ---- one-time weight convert+transpose to bf16 [N][K] ----
    convT_kernel<<<cdiv(NUM_FEATS * DIM, 256), 256, 0, stream>>>(w1a, w1aT, NUM_FEATS, DIM);
    convT_kernel<<<cdiv(DIM * DIM, 256), 256, 0, stream>>>(w1b, w1bT, DIM, DIM);
    convT_kernel<<<cdiv(DIM * DIM, 256), 256, 0, stream>>>(w2a, w2aT, DIM, DIM);
    convT_kernel<<<cdiv(DIM * DIM, 256), 256, 0, stream>>>(w2b, w2bT, DIM, DIM);
    convT_kernel<<<cdiv(DIM * DIM, 256), 256, 0, stream>>>(w3a, w3aT, DIM, DIM);
    convT_kernel<<<cdiv(DIM * DIM, 256), 256, 0, stream>>>(w3b, w3bT, DIM, DIM);
    convT_kernel<<<cdiv(DIM * DIM, 256), 256, 0, stream>>>(fw1, fw1T, DIM, DIM);
    convT_kernel<<<cdiv(DIM * NUM_CLASSES, 256), 256, 0, stream>>>(fw2, fw2T, DIM, NUM_CLASSES);

    const dim3 gemm_blk(128);
    const int  mt50k = cdiv(N_NODES, 32);   // 1563

    // ================= GIN layer 1 (input dim 128) =================
    {
        int D = NUM_FEATS;
        copy4_kernel<<<cdiv(N_NODES * D / 4, 256), 256, 0, stream>>>(x, zsum, N_NODES * D / 4);
        edge_agg_kernel<<<cdiv(N_EDGES * (D / 4), 256), 256, 0, stream>>>(x, src, dst, zsum, N_EDGES, D);
        f2bf_kernel<<<cdiv(N_NODES * D / 4, 256), 256, 0, stream>>>(zsum, abf, N_NODES * D / 4);
        gemm_wmma_kernel<<<dim3(cdiv(DIM, 128), mt50k), gemm_blk, 0, stream>>>(
            abf, w1aT, b1a, t1bf, N_NODES, DIM, D, 1, 1);
        gemm_wmma_kernel<<<dim3(cdiv(DIM, 128), mt50k), gemm_blk, 0, stream>>>(
            t1bf, w1bT, b1b, zsum, N_NODES, DIM, DIM, 1, 0);
        zero_kernel<<<2, 256, 0, stream>>>(stats, 512);
        bn_stats_kernel<<<cdiv(N_NODES, 128), 256, 0, stream>>>(zsum, stats, N_NODES);
        bn_apply_kernel<<<N_NODES, 256, 0, stream>>>(zsum, stats, g1, be1, hbuf, N_NODES);
    }
    // ================= GIN layers 2 & 3 (dim 256) =================
    const unsigned short* waT[2] = { w2aT, w3aT };
    const unsigned short* wbT[2] = { w2bT, w3bT };
    const float* ba_[2] = { b2a, b3a };
    const float* bb_[2] = { b2b, b3b };
    const float* gg_[2] = { g2, g3 };
    const float* bt_[2] = { be2, be3 };
    for (int l = 0; l < 2; ++l) {
        copy4_kernel<<<cdiv((int)(ND / 4), 256), 256, 0, stream>>>(hbuf, zsum, (int)(ND / 4));
        edge_agg_kernel<<<cdiv(N_EDGES * (DIM / 4), 256), 256, 0, stream>>>(hbuf, src, dst, zsum, N_EDGES, DIM);
        f2bf_kernel<<<cdiv((int)(ND / 4), 256), 256, 0, stream>>>(zsum, abf, (int)(ND / 4));
        gemm_wmma_kernel<<<dim3(cdiv(DIM, 128), mt50k), gemm_blk, 0, stream>>>(
            abf, waT[l], ba_[l], t1bf, N_NODES, DIM, DIM, 1, 1);
        gemm_wmma_kernel<<<dim3(cdiv(DIM, 128), mt50k), gemm_blk, 0, stream>>>(
            t1bf, wbT[l], bb_[l], zsum, N_NODES, DIM, DIM, 1, 0);
        zero_kernel<<<2, 256, 0, stream>>>(stats, 512);
        bn_stats_kernel<<<cdiv(N_NODES, 128), 256, 0, stream>>>(zsum, stats, N_NODES);
        bn_apply_kernel<<<N_NODES, 256, 0, stream>>>(zsum, stats, gg_[l], bt_[l], hbuf, N_NODES);
    }

    // ================= readout =================
    zero_kernel<<<cdiv(NUM_GRAPHS * DIM, 256), 256, 0, stream>>>(pooled, NUM_GRAPHS * DIM);
    pool_kernel<<<N_NODES, 256, 0, stream>>>(hbuf, batch, pooled);
    f2bf_kernel<<<cdiv(NUM_GRAPHS * DIM / 4, 256), 256, 0, stream>>>(pooled, abf, NUM_GRAPHS * DIM / 4);
    gemm_wmma_kernel<<<dim3(cdiv(DIM, 128), cdiv(NUM_GRAPHS, 32)), gemm_blk, 0, stream>>>(
        abf, fw1T, fb1, headhbf, NUM_GRAPHS, DIM, DIM, 1, 1);
    gemm_wmma_kernel<<<dim3(1, cdiv(NUM_GRAPHS, 32)), gemm_blk, 0, stream>>>(
        headhbf, fw2T, fb2, (float*)d_out, NUM_GRAPHS, NUM_CLASSES, DIM, 0, 0);
}
